// SemanticRVQ_88399016886958
// MI455X (gfx1250) — matrix operand order repaired and checked by reading, not compiled, not run on hardware
//
#include <hip/hip_runtime.h>
#include <hip/hip_bf16.h>

typedef __attribute__((ext_vector_type(16))) __bf16 v16bf;
typedef __attribute__((ext_vector_type(8)))  float  v8f;
typedef __attribute__((ext_vector_type(4)))  int    v4i;

#define NUM_LAYERS 3
#define K_CB   2048
#define DIM    256
#define NROWS  32768            // B*T
#define NCHUNK 128              // K_CB / 16 codewords per WMMA tile
#define KCH    8                // DIM / 32 k-slices per WMMA chain
#define WAVES  4
#define ROWS_PB (WAVES * 16)    // 64 rows per block
#define CHUNK_ELEMS (KCH * 512) // 4096 bf16 = 8 KB per 16-codeword B tile
#define BFRAG_ELEMS ((size_t)NUM_LAYERS * NCHUNK * CHUNK_ELEMS)

// ---------------------------------------------------------------------------
// Pre-pass 1: swizzle fp32 codebook into bf16 WMMA B-fragment order.
// Fragment (l, n0, kc): 32 lanes x 16 bf16. Lane holds column n = lane&15
// (codeword n0*16 + n); elements are 16 contiguous K, half-waves split
// K 0-15 / 16-31 of the 32-wide k slice (contiguous order, same as A).
// ---------------------------------------------------------------------------
__global__ void rvq_swizzle_kernel(const float* __restrict__ cb,
                                   __bf16* __restrict__ bfrag) {
    int tid  = blockIdx.x * blockDim.x + threadIdx.x;   // < BFRAG_ELEMS
    int e    = tid & 15;
    int lane = (tid >> 4) & 31;
    int kc   = (tid >> 9) & 7;
    int n0   = (tid >> 12) & 127;
    int l    = tid >> 19;
    int cw   = n0 * 16 + (lane & 15);
    int k    = kc * 32 + e + ((lane >> 4) << 4);
    float v  = cb[((size_t)(l * K_CB + cw)) * DIM + k];
    bfrag[tid] = (__bf16)v;
}

// ---------------------------------------------------------------------------
// Pre-pass 2: ||w||^2 per codeword (fp32).
// ---------------------------------------------------------------------------
__global__ void rvq_wsq_kernel(const float* __restrict__ cb,
                               float* __restrict__ wsq) {
    int tid = blockIdx.x * blockDim.x + threadIdx.x;    // < NUM_LAYERS*K_CB
    const float4* w = (const float4*)(cb + (size_t)tid * DIM);
    float s = 0.f;
    #pragma unroll 4
    for (int i = 0; i < DIM / 4; ++i) {
        float4 v = w[i];
        s += v.x * v.x + v.y * v.y + v.z * v.z + v.w * v.w;
    }
    wsq[tid] = s;
}

// ---------------------------------------------------------------------------
// Async staging of one 8 KB B tile into LDS (128 threads cooperatively).
// Prefers CDNA5 GLOBAL_LOAD_ASYNC_TO_LDS_B128 (ASYNCcnt), falls back to a
// plain load + ds_store pipeline if the builtin is unavailable.
// ---------------------------------------------------------------------------
__device__ __forceinline__ void stage_btile(const __bf16* __restrict__ src,
                                            __bf16* dst, int tid) {
#if __has_builtin(__builtin_amdgcn_global_load_async_to_lds_b128)
    #pragma unroll
    for (int t = 0; t < 4; ++t) {
        int eo = (tid + t * 128) * 8;          // 8 bf16 = 16 B per op
        __builtin_amdgcn_global_load_async_to_lds_b128(
            (v4i*)(src + eo), (v4i*)(dst + eo), 0, 0);
    }
#else
    #pragma unroll
    for (int t = 0; t < 4; ++t) {
        int eo = (tid + t * 128) * 8;
        *(float4*)(dst + eo) = *(const float4*)(src + eo);
    }
#endif
}

__device__ __forceinline__ void wait_async_done() {
#if __has_builtin(__builtin_amdgcn_s_wait_asynccnt)
    __builtin_amdgcn_s_wait_asynccnt(0);
#else
    asm volatile("s_wait_asynccnt 0" ::: "memory");
#endif
}

// ---------------------------------------------------------------------------
// Main fused RVQ kernel: 4 waves per block, 16 rows per wave.
// B tiles double-buffered in LDS and shared by all 4 waves (4x less L2
// traffic); residual exact fp32 in LDS with rotate-by-5 bank swizzle.
// Two independent WMMA accumulator chains per wave for ILP.
// ---------------------------------------------------------------------------
__global__ __launch_bounds__(128)
__attribute__((amdgpu_waves_per_eu(4)))
void rvq_main_kernel(
    const float*  __restrict__ h,
    const float*  __restrict__ cb,
    const __bf16* __restrict__ bfrag,
    const float*  __restrict__ wsq,
    float* __restrict__ out_q,
    float* __restrict__ out_idx) {

    __shared__ float res[ROWS_PB * DIM];                     // 64 KB
    __shared__ alignas(64) __bf16 btile[2][CHUNK_ELEMS];     // 16 KB (dbl buf)

    const int tid  = threadIdx.x;
    const int wave = tid >> 5;
    const int lane = tid & 31;
    const int half = lane >> 4;                     // 0 | 1
    const int mr   = lane & 15;
    const int row0 = blockIdx.x * ROWS_PB;          // global row base of block
    const int rowW = row0 + wave * 16;              // global row base of wave
    const int mW   = wave * 16;                     // LDS row base of wave

    // Load h -> swizzled LDS residual (coalesced).
    #pragma unroll 4
    for (int m = 0; m < ROWS_PB; ++m) {
        #pragma unroll
        for (int j = 0; j < 2; ++j) {
            int k = tid + 128 * j;
            res[m * DIM + ((k + 5 * m) & 255)] = h[(size_t)(row0 + m) * DIM + k];
        }
    }
    __syncthreads();

    for (int l = 0; l < NUM_LAYERS; ++l) {
        // ---- Build A fragments (16 contiguous K per half-wave, matching B), fp32->bf16.
        v16bf A[KCH];
        #pragma unroll
        for (int kc = 0; kc < KCH; ++kc) {
            #pragma unroll
            for (int e = 0; e < 16; ++e) {
                int k = kc * 32 + e + half * 16;
                int mrow = mW + mr;
                A[kc][e] = (__bf16)res[mrow * DIM + ((k + 5 * mrow) & 255)];
            }
        }

        float bv[8];
        int   bi[8];
        #pragma unroll
        for (int r = 0; r < 8; ++r) { bv[r] = __builtin_inff(); bi[r] = 0; }

        const __bf16* bl = bfrag + (size_t)l * NCHUNK * CHUNK_ELEMS;
        const float*  wl = wsq + l * K_CB;

        // Preload chunk 0 into buffer 0.
        stage_btile(bl, &btile[0][0], tid);
        wait_async_done();
        __syncthreads();

        // ---- Scan all 2048 codewords: 128 chunks x (8 WMMA over K=256).
        for (int n = 0; n < NCHUNK; ++n) {
            const __bf16* nxt = bl + (size_t)(n + 1) * CHUNK_ELEMS;
            if (n + 1 < NCHUNK)
                stage_btile(nxt, &btile[(n + 1) & 1][0], tid);
            __builtin_prefetch(nxt + CHUNK_ELEMS, 0, 3);   // n+2 into caches

            float w = wl[n * 16 + mr];                     // ||w_k||^2
            const __bf16* bp = &btile[n & 1][lane * 16];
            v8f C0 = {0.f, 0.f, 0.f, 0.f, 0.f, 0.f, 0.f, 0.f};
            v8f C1 = {0.f, 0.f, 0.f, 0.f, 0.f, 0.f, 0.f, 0.f};
            #pragma unroll
            for (int kc = 0; kc < KCH; kc += 2) {
                v16bf B0 = *(const v16bf*)(bp + kc * 512);
                v16bf B1 = *(const v16bf*)(bp + (kc + 1) * 512);
                C0 = __builtin_amdgcn_wmma_f32_16x16x32_bf16(
                         false, A[kc], false, B0, (short)0, C0, false, false);
                C1 = __builtin_amdgcn_wmma_f32_16x16x32_bf16(
                         false, A[kc + 1], false, B1, (short)0, C1, false, false);
            }
            int kidx = n * 16 + mr;
            #pragma unroll
            for (int r = 0; r < 8; ++r) {
                float s = C0[r] + C1[r];
                float d = fmaf(-2.0f, s, w);  // ||r||^2 const over k -> dropped
                if (d < bv[r]) { bv[r] = d; bi[r] = kidx; }
            }

            wait_async_done();
            __syncthreads();
        }

        // ---- Argmin butterfly within each 16-lane half (first-index ties).
        #pragma unroll
        for (int r = 0; r < 8; ++r) {
            #pragma unroll
            for (int off = 8; off >= 1; off >>= 1) {
                float ov = __shfl_xor(bv[r], off, 32);
                int   oi = __shfl_xor(bi[r], off, 32);
                if (ov < bv[r] || (ov == bv[r] && oi < bi[r])) {
                    bv[r] = ov; bi[r] = oi;
                }
            }
        }

        // Emit indices: lane 0 -> rows 0..7, lane 16 -> rows 8..15 of wave.
        if (mr == 0) {
            #pragma unroll
            for (int r = 0; r < 8; ++r)
                out_idx[(size_t)l * NROWS + rowW + half * 8 + r] = (float)bi[r];
        }

        // Broadcast the 16 winners to every lane of the wave.
        int idxAll[16];
        #pragma unroll
        for (int m = 0; m < 16; ++m)
            idxAll[m] = __shfl(bi[m & 7], (m < 8) ? 0 : 16, 32);

        // ---- Exact fp32 residual update from the original codebook.
        const float* cbl = cb + (size_t)l * K_CB * DIM;
        #pragma unroll
        for (int m = 0; m < 16; ++m) {
            const float* wrow = cbl + (size_t)idxAll[m] * DIM;
            int mrow = mW + m;
            #pragma unroll 2
            for (int j = 0; j < 8; ++j) {
                int k = lane + 32 * j;
                res[mrow * DIM + ((k + 5 * mrow) & 255)] -= wrow[k];
            }
        }
        __syncthreads();
    }

    // quantized_total = h - residual_final  (exact fp32).
    #pragma unroll 4
    for (int m = 0; m < ROWS_PB; ++m) {
        #pragma unroll
        for (int j = 0; j < 2; ++j) {
            int k = tid + 128 * j;
            size_t g = (size_t)(row0 + m) * DIM + k;
            out_q[g] = h[g] - res[m * DIM + ((k + 5 * m) & 255)];
        }
    }
}

// ---------------------------------------------------------------------------
extern "C" void kernel_launch(void* const* d_in, const int* in_sizes, int n_in,
                              void* d_out, int out_size, void* d_ws, size_t ws_size,
                              hipStream_t stream) {
    const float* h  = (const float*)d_in[0];   // (16, 2048, 256) fp32
    const float* cb = (const float*)d_in[1];   // (3, 2048, 256) fp32

    __bf16* bfrag = (__bf16*)d_ws;
    float*  wsqp  = (float*)((char*)d_ws + BFRAG_ELEMS * sizeof(__bf16));

    float* out_q   = (float*)d_out;                       // (16,2048,256)
    float* out_idx = out_q + (size_t)NROWS * DIM;         // (3,16,2048)

    rvq_swizzle_kernel<<<(int)(BFRAG_ELEMS / 256), 256, 0, stream>>>(cb, bfrag);
    rvq_wsq_kernel<<<(NUM_LAYERS * K_CB) / 256, 256, 0, stream>>>(cb, wsqp);
    rvq_main_kernel<<<NROWS / ROWS_PB, 128, 0, stream>>>(h, cb, bfrag, wsqp,
                                                         out_q, out_idx);
}
